// peftCLIP_59536836657841
// MI455X (gfx1250) — matrix operand order repaired and checked by reading, not compiled, not run on hardware
//
#include <hip/hip_runtime.h>
#include <math.h>

// ---------------------------------------------------------------------------
// CLIP ViT-B/16 forward for gfx1250 (MI455X).
// f16 WMMA GEMMs (v_wmma_f32_16x16x32_f16), fp32 residual stream + softmax/LN.
// GEMM: 128x128x32 block tile, 8 waves, 8 WMMA per wave per K-step.
// ---------------------------------------------------------------------------

typedef __attribute__((ext_vector_type(16))) _Float16 v16h;
typedef __attribute__((ext_vector_type(2)))  _Float16 v2h;
typedef __attribute__((ext_vector_type(8)))  float    v8f;

#define TB   256          // threads per block (8 waves, wave32)
#define LDT  40           // padded LDS row stride (halfs) for 32-wide K tile

// ------------------------- generic WMMA GEMM -------------------------------
// C[M,N] = act( A[M,K] @ W[N,K]^T + bias ) + res ; store f32 and/or f16.
// Block tile 128x128, K step 32. Wave (wm,wn): wm in 0..3 (32-row strip),
// wn in 0..1 (64-col strip) -> 2 A-frags x 4 B-frags -> 8 WMMA / K-step.
__global__ __launch_bounds__(TB) void k_gemm(
    const _Float16* __restrict__ A, const _Float16* __restrict__ W,
    const float* __restrict__ bias, const float* __restrict__ res,
    float* __restrict__ Cf, _Float16* __restrict__ Ch,
    int M, int N, int K, int act)
{
    __shared__ _Float16 sA[128 * LDT];
    __shared__ _Float16 sB[128 * LDT];

    const int tid  = threadIdx.x;
    const int wave = tid >> 5;
    const int lane = tid & 31;
    const int m0 = blockIdx.y * 128;
    const int n0 = blockIdx.x * 128;
    const int wm = wave & 3;      // 0..3 : 32-row strip
    const int wn = wave >> 2;     // 0..1 : 64-col strip

    v8f acc[2][4];
#pragma unroll
    for (int i = 0; i < 2; ++i)
#pragma unroll
        for (int j = 0; j < 4; ++j) acc[i][j] = (v8f){};

    // cooperative tile load mapping: 128 rows x 32 halfs; 32B (16 halfs)/thread
    const int lrow = tid >> 1;          // 0..127
    const int lcol = (tid & 1) * 16;    // 0 or 16 (halfs)

    const int lm = lane & 15;
    const int hi = lane >> 4;           // 0 or 1

    for (int kk = 0; kk < K; kk += 32) {
        // ---- stage tiles into LDS ----
        {
            int grow = m0 + lrow;
            uint4 ld0 = {0u, 0u, 0u, 0u}, ld1 = {0u, 0u, 0u, 0u};
            if (grow < M) {
                const uint4* gp = (const uint4*)(A + (size_t)grow * K + kk + lcol);
                ld0 = gp[0];
                ld1 = gp[1];
            }
            uint4* sp = (uint4*)(sA + lrow * LDT + lcol);
            sp[0] = ld0;
            sp[1] = ld1;
        }
        {
            int grow = n0 + lrow;
            uint4 ld0 = {0u, 0u, 0u, 0u}, ld1 = {0u, 0u, 0u, 0u};
            if (grow < N) {
                const uint4* gp = (const uint4*)(W + (size_t)grow * K + kk + lcol);
                ld0 = gp[0];
                ld1 = gp[1];
            }
            uint4* sp = (uint4*)(sB + lrow * LDT + lcol);
            sp[0] = ld0;
            sp[1] = ld1;
        }
        // prefetch next K tile into caches while we compute this one
        if (kk + 32 < K) {
            int grow = m0 + lrow;
            if (grow < M) __builtin_prefetch(A + (size_t)grow * K + kk + 32 + lcol, 0, 3);
            int grow2 = n0 + lrow;
            if (grow2 < N) __builtin_prefetch(W + (size_t)grow2 * K + kk + 32 + lcol, 0, 3);
        }
        __syncthreads();

        // ---- A fragments: 16x32 f16; lanes 0-15 K-base 0, lanes 16-31 K+8 ----
        v16h af[2];
#pragma unroll
        for (int i = 0; i < 2; ++i) {
            const _Float16* arow = sA + (wm * 32 + i * 16 + lm) * LDT;
            const int koff = hi * 8;
#pragma unroll
            for (int r = 0; r < 8; ++r) {
                int k0 = ((r < 4) ? 2 * r : 2 * r + 8) + koff;
                v2h p = *(const v2h*)(arow + k0);
                af[i][2 * r]     = p[0];
                af[i][2 * r + 1] = p[1];
            }
        }
        // ---- B fragments: 32x16 f16; lane = N col; lanes 16-31 K+8 ----
        v16h bf[4];
#pragma unroll
        for (int j = 0; j < 4; ++j) {
            const _Float16* brow = sB + (wn * 64 + j * 16 + lm) * LDT;
            const int koff = hi * 8;
#pragma unroll
            for (int r = 0; r < 8; ++r) {
                int k0 = ((r < 4) ? 2 * r : 2 * r + 8) + koff;
                v2h p = *(const v2h*)(brow + k0);
                bf[j][2 * r]     = p[0];
                bf[j][2 * r + 1] = p[1];
            }
        }

#pragma unroll
        for (int i = 0; i < 2; ++i)
#pragma unroll
            for (int j = 0; j < 4; ++j)
                acc[i][j] = __builtin_amdgcn_wmma_f32_16x16x32_f16(
                    false, af[i], false, bf[j], (short)0, acc[i][j], false, false);
        __syncthreads();
    }

    // ---- epilogue: C/D layout lane=N col, VGPR r = M row (+8 for lanes 16-31)
    const int moff = hi * 8;
#pragma unroll
    for (int i = 0; i < 2; ++i) {
#pragma unroll
        for (int r = 0; r < 8; ++r) {
            int row = m0 + wm * 32 + i * 16 + moff + r;
            if (row >= M) continue;
#pragma unroll
            for (int j = 0; j < 4; ++j) {
                int col = n0 + wn * 64 + j * 16 + lm;
                float val = acc[i][j][r];
                if (bias) val += bias[col];
                if (act == 1) val = 0.5f * val * (1.0f + erff(val * 0.70710678118654752f));
                size_t off = (size_t)row * N + col;
                if (res) val += res[off];
                if (Cf)  Cf[off] = val;
                if (Ch)  Ch[off] = (_Float16)val;
            }
        }
    }
}

// ------------------------- elementwise helpers -----------------------------

__global__ void k_cvt(const float* __restrict__ in, _Float16* __restrict__ out,
                      long long n)
{
    long long i = (long long)blockIdx.x * TB + threadIdx.x;
    if (i < n) out[i] = (_Float16)in[i];
}

// proj is [D=768, E=512] (x @ proj) -> store transposed [E, D] f16 for GEMM.
__global__ void k_cvt_transpose(const float* __restrict__ in, _Float16* __restrict__ out,
                                int n)   // n = 768*512
{
    int i = blockIdx.x * TB + threadIdx.x;
    if (i >= n) return;
    int d = i / 512, e = i % 512;
    out[(size_t)e * 768 + d] = (_Float16)in[i];
}

// im2col: A[b*196 + gy*14+gx][c*256 + py*16 + px] = image[b][c][gy*16+py][gx*16+px]
__global__ void k_im2col(const float* __restrict__ img, _Float16* __restrict__ out,
                         int total)   // total = 6272*768
{
    int i = blockIdx.x * TB + threadIdx.x;
    if (i >= total) return;
    int col = i % 768;
    int r   = i / 768;
    int b = r / 196, pidx = r % 196;
    int gy = pidx / 14, gx = pidx % 14;
    int c = col >> 8, py = (col >> 4) & 15, px = col & 15;
    float v = img[(((size_t)b * 3 + c) * 224 + gy * 16 + py) * 224 + gx * 16 + px];
    out[i] = (_Float16)v;
}

// X[b][s][d] = (s==0 ? cls[d] : Xp[b][s-1][d]) + pos[s][d]
__global__ void k_assemble(const float* __restrict__ xp, const float* __restrict__ cls,
                           const float* __restrict__ pos, float* __restrict__ X,
                           int total)   // total = 32*197*768
{
    int i = blockIdx.x * TB + threadIdx.x;
    if (i >= total) return;
    int d = i % 768;
    int r = i / 768;
    int b = r / 197, s = r % 197;
    float v = (s == 0) ? cls[d] : xp[((size_t)b * 196 + (s - 1)) * 768 + d];
    X[i] = v + pos[(size_t)s * 768 + d];
}

// LayerNorm over D=768; one block per row; input row stride parameterized
// (768 for dense, 197*768 to pick out the CLS token rows). Writes compact rows.
__global__ __launch_bounds__(TB) void k_layernorm(
    const float* __restrict__ x, const float* __restrict__ g,
    const float* __restrict__ bta, float* __restrict__ of,
    _Float16* __restrict__ oh, long long rstride)
{
    __shared__ float s1[TB];
    __shared__ float s2[TB];
    const float* row = x + (size_t)blockIdx.x * rstride;
    int t = threadIdx.x;
    float a0 = row[t], a1 = row[t + 256], a2 = row[t + 512];
    s1[t] = a0 + a1 + a2;
    s2[t] = a0 * a0 + a1 * a1 + a2 * a2;
    __syncthreads();
    for (int s = 128; s > 0; s >>= 1) {
        if (t < s) { s1[t] += s1[t + s]; s2[t] += s2[t + s]; }
        __syncthreads();
    }
    float mean = s1[0] * (1.0f / 768.0f);
    float var  = s2[0] * (1.0f / 768.0f) - mean * mean;
    float inv  = rsqrtf(var + 1e-5f);
    size_t ob = (size_t)blockIdx.x * 768;
#pragma unroll
    for (int j = 0; j < 3; ++j) {
        int d = t + j * 256;
        float a = (j == 0) ? a0 : (j == 1) ? a1 : a2;
        float y = (a - mean) * inv * g[d] + bta[d];
        if (of) of[ob + d] = y;
        if (oh) oh[ob + d] = (_Float16)y;
    }
}

// Attention: one block per (query s, b*H+h). qkv f32 [B*SEQ, 3*768].
__global__ __launch_bounds__(TB) void k_attn(const float* __restrict__ qkv,
                                             _Float16* __restrict__ attn)
{
    __shared__ float sq[64];
    __shared__ float sp[TB];
    __shared__ float sr[TB];
    const int s  = blockIdx.x;
    const int bh = blockIdx.y;
    const int b = bh / 12, h = bh % 12;
    const int t = threadIdx.x;
    const size_t row3 = 3 * 768;

    if (t < 64) sq[t] = qkv[((size_t)(b * 197 + s)) * row3 + h * 64 + t] * 0.125f;
    __syncthreads();

    float sc = -1e30f;
    if (t < 197) {
        const float* krow = qkv + ((size_t)(b * 197 + t)) * row3 + 768 + h * 64;
        float acc = 0.f;
#pragma unroll 4
        for (int d = 0; d < 64; ++d) acc += sq[d] * krow[d];
        sc = acc;
    }
    sp[t] = sc;
    sr[t] = sc;
    __syncthreads();
    for (int st = 128; st > 0; st >>= 1) {
        if (t < st) sr[t] = fmaxf(sr[t], sr[t + st]);
        __syncthreads();
    }
    float mx = sr[0];
    __syncthreads();
    float e = (t < 197) ? __expf(sc - mx) : 0.f;
    sr[t] = e;
    __syncthreads();
    for (int st = 128; st > 0; st >>= 1) {
        if (t < st) sr[t] += sr[t + st];
        __syncthreads();
    }
    float invsum = 1.0f / sr[0];
    __syncthreads();
    sp[t] = e * invsum;
    __syncthreads();

    if (t < 64) {
        float o = 0.f;
        const float* vbase = qkv + 1536 + h * 64 + t;
        for (int j = 0; j < 197; ++j)
            o += sp[j] * vbase[((size_t)(b * 197 + j)) * row3];
        attn[((size_t)(b * 197 + s)) * 768 + h * 64 + t] = (_Float16)o;
    }
}

// L2-normalize rows of feats [B,512] -> d_out.
__global__ __launch_bounds__(TB) void k_l2norm(const float* __restrict__ f,
                                               float* __restrict__ out)
{
    __shared__ float sr[TB];
    int b = blockIdx.x, t = threadIdx.x;
    float a0 = f[(size_t)b * 512 + t];
    float a1 = f[(size_t)b * 512 + t + 256];
    sr[t] = a0 * a0 + a1 * a1;
    __syncthreads();
    for (int s = 128; s > 0; s >>= 1) {
        if (t < s) sr[t] += sr[t + s];
        __syncthreads();
    }
    float inv = rsqrtf(sr[0]);
    out[(size_t)b * 512 + t]       = a0 * inv;
    out[(size_t)b * 512 + t + 256] = a1 * inv;
}

// ---------------------------------------------------------------------------

extern "C" void kernel_launch(void* const* d_in, const int* in_sizes, int n_in,
                              void* d_out, int out_size, void* d_ws, size_t ws_size,
                              hipStream_t stream)
{
    const float* image   = (const float*)d_in[0];
    const float* conv_w  = (const float*)d_in[1];
    const float* cls_emb = (const float*)d_in[2];
    const float* pos_emb = (const float*)d_in[3];
    const float* lnprew  = (const float*)d_in[4];
    const float* lnpreb  = (const float*)d_in[5];
    const float* ln1w    = (const float*)d_in[6];
    const float* ln1b    = (const float*)d_in[7];
    const float* qkvw    = (const float*)d_in[8];
    const float* qkvb    = (const float*)d_in[9];
    const float* outw    = (const float*)d_in[10];
    const float* outb    = (const float*)d_in[11];
    const float* ln2w    = (const float*)d_in[12];
    const float* ln2b    = (const float*)d_in[13];
    const float* fcw     = (const float*)d_in[14];
    const float* fcb     = (const float*)d_in[15];
    const float* prw     = (const float*)d_in[16];
    const float* prb     = (const float*)d_in[17];
    const float* lnpow   = (const float*)d_in[18];
    const float* lnpob   = (const float*)d_in[19];
    const float* proj    = (const float*)d_in[20];

    const int B = 32, SEQ = 197, D = 768, FF = 3072, L = 12, E = 512;
    const int NT = B * SEQ;   // 6304 tokens
    const int NP = B * 196;   // 6272 patch tokens

    // ---- workspace bump allocator ----
    char* wsb = (char*)d_ws;
    size_t off = 0;
    auto alloc = [&](size_t bytes) -> void* {
        void* p = wsb + off;
        off = (off + bytes + 255) & ~(size_t)255;
        return p;
    };

    // f16 weight copies
    _Float16* conv_h = (_Float16*)alloc((size_t)D * D * 2);
    _Float16* qkv_h  = (_Float16*)alloc((size_t)L * 3 * D * D * 2);
    _Float16* out_h  = (_Float16*)alloc((size_t)L * D * D * 2);
    _Float16* fc_h   = (_Float16*)alloc((size_t)L * FF * D * 2);
    _Float16* pr_h   = (_Float16*)alloc((size_t)L * D * FF * 2);
    _Float16* projT  = (_Float16*)alloc((size_t)E * D * 2);

    // activations
    _Float16* Ap    = (_Float16*)alloc((size_t)NP * D * 2);   // im2col patches
    float*    Xp    = (float*)   alloc((size_t)NP * D * 4);   // patch embeddings
    float*    Xa    = (float*)   alloc((size_t)NT * D * 4);   // pre-ln_pre
    float*    X     = (float*)   alloc((size_t)NT * D * 4);   // residual stream
    _Float16* Hh    = (_Float16*)alloc((size_t)NT * D * 2);   // LN output (f16)
    float*    QKV   = (float*)   alloc((size_t)NT * 3 * D * 4);
    _Float16* At    = (_Float16*)alloc((size_t)NT * D * 2);   // attention output
    _Float16* Gh    = (_Float16*)alloc((size_t)NT * FF * 2);  // gelu(fc) output
    _Float16* ClsH  = (_Float16*)alloc((size_t)B * D * 2);
    float*    Feat  = (float*)   alloc((size_t)B * E * 4);
    (void)ws_size; (void)n_in; (void)in_sizes; (void)out_size;

    auto cvt = [&](const float* src, _Float16* dst, long long n) {
        k_cvt<<<dim3((unsigned)((n + TB - 1) / TB)), dim3(TB), 0, stream>>>(src, dst, n);
    };
    auto gemm = [&](const _Float16* A, const _Float16* W, const float* bias,
                    const float* res, float* Cf, _Float16* Ch,
                    int M, int N, int K, int act) {
        dim3 g(N / 128, (M + 127) / 128);
        k_gemm<<<g, dim3(TB), 0, stream>>>(A, W, bias, res, Cf, Ch, M, N, K, act);
    };

    // ---- weight precision conversion (read fp32 once; live in L2 as f16) ----
    cvt(conv_w, conv_h, (long long)D * D);
    cvt(qkvw,   qkv_h,  (long long)L * 3 * D * D);
    cvt(outw,   out_h,  (long long)L * D * D);
    cvt(fcw,    fc_h,   (long long)L * FF * D);
    cvt(prw,    pr_h,   (long long)L * D * FF);
    k_cvt_transpose<<<dim3((D * E + TB - 1) / TB), dim3(TB), 0, stream>>>(proj, projT, D * E);

    // ---- patchify: im2col + GEMM [6272,768] x [768,768]^T ----
    k_im2col<<<dim3((NP * D + TB - 1) / TB), dim3(TB), 0, stream>>>(image, Ap, NP * D);
    gemm(Ap, conv_h, nullptr, nullptr, Xp, nullptr, NP, D, D, 0);

    // ---- cls + pos, ln_pre ----
    k_assemble<<<dim3((NT * D + TB - 1) / TB), dim3(TB), 0, stream>>>(Xp, cls_emb, pos_emb, Xa, NT * D);
    k_layernorm<<<dim3(NT), dim3(TB), 0, stream>>>(Xa, lnprew, lnpreb, X, nullptr, 768);

    // ---- transformer layers ----
    for (int l = 0; l < L; ++l) {
        const _Float16* Wq = qkv_h + (size_t)l * 3 * D * D;
        const _Float16* Wo = out_h + (size_t)l * D * D;
        const _Float16* Wf = fc_h  + (size_t)l * FF * D;
        const _Float16* Wp = pr_h  + (size_t)l * D * FF;

        // h = ln1(x)  -> f16
        k_layernorm<<<dim3(NT), dim3(TB), 0, stream>>>(X, ln1w + l * D, ln1b + l * D,
                                                       nullptr, Hh, 768);
        // qkv = h @ Wq^T + bq
        gemm(Hh, Wq, qkvb + (size_t)l * 3 * D, nullptr, QKV, nullptr, NT, 3 * D, D, 0);
        // attention
        k_attn<<<dim3(SEQ, B * 12), dim3(TB), 0, stream>>>(QKV, At);
        // x = x + attn @ Wo^T + bo
        gemm(At, Wo, outb + (size_t)l * D, X, X, nullptr, NT, D, D, 0);
        // h = ln2(x) -> f16
        k_layernorm<<<dim3(NT), dim3(TB), 0, stream>>>(X, ln2w + l * D, ln2b + l * D,
                                                       nullptr, Hh, 768);
        // g = gelu(h @ Wf^T + bf) -> f16
        gemm(Hh, Wf, fcb + (size_t)l * FF, nullptr, nullptr, Gh, NT, FF, D, 1);
        // x = x + g @ Wp^T + bp
        gemm(Gh, Wp, prb + (size_t)l * D, X, X, nullptr, NT, D, FF, 0);
    }

    // ---- ln_post on CLS rows (input row stride SEQ*D), project, normalize ----
    k_layernorm<<<dim3(B), dim3(TB), 0, stream>>>(X, lnpow, lnpob, nullptr, ClsH,
                                                  (long long)SEQ * D);
    gemm(ClsH, projT, nullptr, nullptr, Feat, nullptr, B, E, D, 0);
    k_l2norm<<<dim3(B), dim3(TB), 0, stream>>>(Feat, (float*)d_out);
}
